// MedicalVisionTransformer_40372692582708
// MI455X (gfx1250) — matrix operand order repaired and checked by reading, not compiled, not run on hardware
//
#include <hip/hip_runtime.h>
#include <hip/hip_bf16.h>
#include <math.h>

// ---- problem constants (match reference) ----
#define L_  12
#define H_  768
#define I_  3072
#define NH_ 12
#define HD_ 64
#define E_  15
#define D_  14
#define R_  29
#define B_  128
#define S_  (R_ + 1)        // 30
#define LR_ 8
#define NE_ (L_ / 2)
#define T_  (B_ * S_)       // 3840 tokens
#define SCAL_ 2.0f          // lora_alpha / lora_r

// ---- WMMA types ----
typedef __attribute__((ext_vector_type(16))) __bf16 v16bf;
typedef __attribute__((ext_vector_type(8)))  float  v8f;

union ABfrag { v16bf v; unsigned int u[8]; };
union CFrag  { v8f v; float f[8]; };

__device__ __forceinline__ unsigned pk_bf16(float a, float b) {
    unsigned ua = __float_as_uint(a); ua += 0x7FFFu + ((ua >> 16) & 1u);
    unsigned ub = __float_as_uint(b); ub += 0x7FFFu + ((ub >> 16) & 1u);
    return (ua >> 16) | (ub & 0xFFFF0000u);
}
__device__ __forceinline__ float gelu_exact(float x) {
    return 0.5f * x * (1.0f + erff(x * 0.70710678118654752f));
}

// ============================================================
// GEMM: out[M,N] = act(A[M,K] @ W[N,K]^T + bias)
// 128x64 block tile, 8 waves in 4x2 grid, each wave owns 32x32
// (2 A-frags x 2 B-frags -> 4 WMMAs per K-step, each frag reused 2x).
// bf16 inputs (converted during LDS staging), f32 accumulation.
// Requires M % 128 == 0, N % 64 == 0, K % 32 == 0.
// ============================================================
template<int ACT>
__global__ __launch_bounds__(256) void gemm_wmma(
    const float* __restrict__ A, const float* __restrict__ W,
    const float* __restrict__ bias, float* __restrict__ out,
    int M, int N, int K)
{
    __shared__ unsigned As[128 * 17];   // 128 rows x 16 bf16-pairs, pad 17
    __shared__ unsigned Wsh[64 * 17];   // 64 rows x 16 bf16-pairs

    const int tid  = threadIdx.x;
    const int m0   = blockIdx.y * 128;
    const int n0   = blockIdx.x * 64;
    const int lane = tid & 31;
    const int wave = tid >> 5;
    const int wm   = wave & 3;          // 0..3 -> 32-row slice
    const int wn   = wave >> 2;         // 0..1 -> 32-col slice
    const int lr   = lane & 15;
    const int hi   = lane >> 4;

    CFrag acc00, acc01, acc10, acc11;
    #pragma unroll
    for (int r = 0; r < 8; ++r) {
        acc00.f[r] = 0.f; acc01.f[r] = 0.f; acc10.f[r] = 0.f; acc11.f[r] = 0.f;
    }

    const int nk = K >> 5;
    for (int kt = 0; kt < nk; ++kt) {
        const int k0 = kt << 5;
        __syncthreads();
        // stage 128x32 A + 64x32 W as packed bf16 pairs: 3072 dwords / 256 thr
        #pragma unroll
        for (int i = 0; i < 12; ++i) {
            int idx = tid + i * 256;
            if (idx < 2048) {
                int row = idx >> 4, cp = idx & 15;
                const float* pa = A + (size_t)(m0 + row) * K + k0 + cp * 2;
                As[row * 17 + cp] = pk_bf16(pa[0], pa[1]);
            } else {
                int j = idx - 2048;
                int row = j >> 4, cp = j & 15;
                const float* pw = W + (size_t)(n0 + row) * K + k0 + cp * 2;
                Wsh[row * 17 + cp] = pk_bf16(pw[0], pw[1]);
            }
        }
        // prefetch next K-tile (global_prefetch_b8): one 128B row-chunk/thread
        if (kt + 1 < nk) {
            int row = tid & 127;
            __builtin_prefetch(A + (size_t)(m0 + row) * K + k0 + 32, 0, 1);
            if (tid < 64)
                __builtin_prefetch(W + (size_t)(n0 + tid) * K + k0 + 32, 0, 1);
        }
        __syncthreads();

        // A fragments: 16x32 bf16, contiguous k order (k = 16*hi + s)
        // lanes 0-15: M rows, K pairs {0..7}; lanes 16-31: {8..15}
        ABfrag a0, a1;
        const int ar0 = wm * 32 + lr;
        #pragma unroll
        for (int j = 0; j < 8; ++j) {
            int p = hi * 8 + j;
            a0.u[j] = As[ar0 * 17 + p];
            a1.u[j] = As[(ar0 + 16) * 17 + p];
        }
        // B fragments (W^T): lane -> column, VGPR r -> K pair (hi picks K half)
        ABfrag b0, b1;
        const int bc0 = wn * 32 + lr;
        #pragma unroll
        for (int r = 0; r < 8; ++r) {
            b0.u[r] = Wsh[bc0 * 17 + hi * 8 + r];
            b1.u[r] = Wsh[(bc0 + 16) * 17 + hi * 8 + r];
        }
        acc00.v = __builtin_amdgcn_wmma_f32_16x16x32_bf16(
            false, a0.v, false, b0.v, (short)0, acc00.v, false, false);
        acc01.v = __builtin_amdgcn_wmma_f32_16x16x32_bf16(
            false, a0.v, false, b1.v, (short)0, acc01.v, false, false);
        acc10.v = __builtin_amdgcn_wmma_f32_16x16x32_bf16(
            false, a1.v, false, b0.v, (short)0, acc10.v, false, false);
        acc11.v = __builtin_amdgcn_wmma_f32_16x16x32_bf16(
            false, a1.v, false, b1.v, (short)0, acc11.v, false, false);
    }

    // store: lane = column, VGPR r = row (hi adds +8)
    const int col0 = n0 + wn * 32 + lr;
    const float bs0 = bias ? bias[col0] : 0.f;
    const float bs1 = bias ? bias[col0 + 16] : 0.f;
    const int rowb = m0 + wm * 32 + hi * 8;
    #pragma unroll
    for (int r = 0; r < 8; ++r) {
        size_t o0 = (size_t)(rowb + r) * N;
        size_t o1 = (size_t)(rowb + 16 + r) * N;
        float x00 = acc00.f[r] + bs0;
        float x01 = acc01.f[r] + bs1;
        float x10 = acc10.f[r] + bs0;
        float x11 = acc11.f[r] + bs1;
        if (ACT == 1) {
            x00 = gelu_exact(x00); x01 = gelu_exact(x01);
            x10 = gelu_exact(x10); x11 = gelu_exact(x11);
        }
        out[o0 + col0]      = x00;
        out[o0 + col0 + 16] = x01;
        out[o1 + col0]      = x10;
        out[o1 + col0 + 16] = x11;
    }
}

// ============================================================
// LayerNorm over last dim (one block per row)
// ============================================================
__global__ __launch_bounds__(256) void layernorm_k(
    const float* __restrict__ in, const float* __restrict__ g,
    const float* __restrict__ b, float* __restrict__ out,
    int W, float eps)
{
    __shared__ float r1[256], r2[256];
    const int row = blockIdx.x;
    const float* x = in + (size_t)row * W;
    float s = 0.f, ss = 0.f;
    for (int i = threadIdx.x; i < W; i += 256) { float v = x[i]; s += v; ss += v * v; }
    r1[threadIdx.x] = s; r2[threadIdx.x] = ss; __syncthreads();
    for (int st = 128; st > 0; st >>= 1) {
        if (threadIdx.x < st) { r1[threadIdx.x] += r1[threadIdx.x + st]; r2[threadIdx.x] += r2[threadIdx.x + st]; }
        __syncthreads();
    }
    const float m   = r1[0] / W;
    const float var = r2[0] / W - m * m;
    const float inv = rsqrtf(var + eps);
    float* o = out + (size_t)row * W;
    for (int i = threadIdx.x; i < W; i += 256) {
        float y = (x[i] - m) * inv;
        if (g) y = y * g[i] + b[i];
        o[i] = y;
    }
}

// ============================================================
// h0 = concat(cls, region_features)
// ============================================================
__global__ void init_h_k(const float* __restrict__ reg, const float* __restrict__ cls,
                         float* __restrict__ h)
{
    int idx = blockIdx.x * 256 + threadIdx.x;
    if (idx >= T_ * H_) return;
    int hh = idx % H_;
    int ts = idx / H_;
    int s = ts % S_, b = ts / S_;
    h[idx] = (s == 0) ? cls[hh] : reg[((size_t)(b * R_ + s - 1)) * H_ + hh];
}

__global__ void add_k(const float* __restrict__ a, const float* __restrict__ b,
                      float* __restrict__ o, int n)
{
    int i = blockIdx.x * 256 + threadIdx.x;
    if (i < n) o[i] = a[i] + b[i];
}

// ============================================================
// Attention: one block per (batch, head). S=30, HD=64.
// ============================================================
__global__ __launch_bounds__(128) void attn_k(
    const float* __restrict__ q, const float* __restrict__ k,
    const float* __restrict__ v, float* __restrict__ ctx)
{
    __shared__ float qs[S_ * HD_], ks[S_ * HD_], vs[S_ * HD_];
    __shared__ float sc[S_ * 32];
    const int blk = blockIdx.x;
    const int b = blk / NH_, hh = blk % NH_;
    const size_t base = ((size_t)b * S_) * H_ + hh * HD_;
    for (int idx = threadIdx.x; idx < S_ * HD_; idx += 128) {
        int s = idx / HD_, d = idx % HD_;
        qs[idx] = q[base + (size_t)s * H_ + d];
        ks[idx] = k[base + (size_t)s * H_ + d];
        vs[idx] = v[base + (size_t)s * H_ + d];
    }
    __syncthreads();
    for (int idx = threadIdx.x; idx < S_ * S_; idx += 128) {
        int i = idx / S_, j = idx % S_;
        float acc = 0.f;
        #pragma unroll 8
        for (int d = 0; d < HD_; ++d) acc += qs[i * HD_ + d] * ks[j * HD_ + d];
        sc[i * 32 + j] = acc * 0.125f;   // 1/sqrt(64)
    }
    __syncthreads();
    if (threadIdx.x < S_) {
        const int i = threadIdx.x;
        float mx = -1e30f;
        for (int j = 0; j < S_; ++j) mx = fmaxf(mx, sc[i * 32 + j]);
        float sum = 0.f;
        for (int j = 0; j < S_; ++j) { float e = expf(sc[i * 32 + j] - mx); sc[i * 32 + j] = e; sum += e; }
        float inv = 1.0f / sum;
        for (int j = 0; j < S_; ++j) sc[i * 32 + j] *= inv;
    }
    __syncthreads();
    for (int idx = threadIdx.x; idx < S_ * HD_; idx += 128) {
        int i = idx / HD_, d = idx % HD_;
        float acc = 0.f;
        for (int j = 0; j < S_; ++j) acc += sc[i * 32 + j] * vs[j * HD_ + d];
        ctx[base + (size_t)i * H_ + d] = acc;
    }
}

// ============================================================
// mask prep: maskf = (mask>0), cnt[d] = sum_r maskf
// ============================================================
__global__ void mask_prep_k(const float* __restrict__ mask,
                            float* __restrict__ maskf, float* __restrict__ cnt)
{
    int t = threadIdx.x;
    if (t < R_ * D_) maskf[t] = (mask[t] > 0.f) ? 1.f : 0.f;
    __syncthreads();
    if (t < D_) {
        float c = 0.f;
        for (int r = 0; r < R_; ++r) c += maskf[r * D_ + t];
        cnt[t] = c;
    }
}

// pooled[b,d,h] = sum_r maskf[r,d]*att[b, 1+r, h] / max(cnt[d],1)
__global__ void pool_k(const float* __restrict__ att, const float* __restrict__ maskf,
                       const float* __restrict__ cnt, float* __restrict__ pooled)
{
    int idx = blockIdx.x * 256 + threadIdx.x;
    if (idx >= B_ * D_ * H_) return;
    int hh = idx % H_;
    int bd = idx / H_;
    int d = bd % D_, b = bd / D_;
    float acc = 0.f;
    for (int r = 0; r < R_; ++r) {
        float m = maskf[r * D_ + d];
        if (m != 0.f) acc += att[((size_t)(b * S_ + 1 + r)) * H_ + hh];
    }
    pooled[idx] = acc / fmaxf(cnt[d], 1.f);
}

// ============================================================
// per-(b,d) classifier: 384-wide MLP + LN + gelu + linear -> flag
// ============================================================
__global__ __launch_bounds__(128) void classify_k(
    const float* __restrict__ pooled, const float* __restrict__ cw1,
    const float* __restrict__ cb1, const float* __restrict__ clg,
    const float* __restrict__ clb, const float* __restrict__ cw2,
    const float* __restrict__ cb2, const float* __restrict__ cnt,
    float* __restrict__ flag, int ei)
{
    const int K2 = H_ / 2;   // 384
    __shared__ float pld[H_];
    __shared__ float h1[H_ / 2];
    __shared__ float red[128];
    const int b = blockIdx.x / D_, d = blockIdx.x % D_;
    const float* pr = pooled + ((size_t)(b * D_ + d)) * H_;
    for (int i = threadIdx.x; i < H_; i += 128) pld[i] = pr[i];
    __syncthreads();
    const size_t wb = ((size_t)(ei * D_ + d)) * K2;
    for (int kk = threadIdx.x; kk < K2; kk += 128) {
        const float* wr = cw1 + (wb + kk) * H_;
        float acc = cb1[wb + kk];
        for (int i = 0; i < H_; ++i) acc += wr[i] * pld[i];
        h1[kk] = acc;
    }
    __syncthreads();
    // LN(384)
    float s = 0.f, ss = 0.f;
    for (int kk = threadIdx.x; kk < K2; kk += 128) { float x = h1[kk]; s += x; ss += x * x; }
    red[threadIdx.x] = s; __syncthreads();
    for (int st = 64; st > 0; st >>= 1) { if (threadIdx.x < st) red[threadIdx.x] += red[threadIdx.x + st]; __syncthreads(); }
    const float mean = red[0] / K2; __syncthreads();
    red[threadIdx.x] = ss; __syncthreads();
    for (int st = 64; st > 0; st >>= 1) { if (threadIdx.x < st) red[threadIdx.x] += red[threadIdx.x + st]; __syncthreads(); }
    const float var = red[0] / K2 - mean * mean;
    const float inv = rsqrtf(var + 1e-5f);
    __syncthreads();
    float part = 0.f;
    for (int kk = threadIdx.x; kk < K2; kk += 128) {
        float x = (h1[kk] - mean) * inv * clg[wb + kk] + clb[wb + kk];
        x = gelu_exact(x);
        part += x * cw2[wb + kk];
    }
    red[threadIdx.x] = part; __syncthreads();
    for (int st = 64; st > 0; st >>= 1) { if (threadIdx.x < st) red[threadIdx.x] += red[threadIdx.x + st]; __syncthreads(); }
    if (threadIdx.x == 0) {
        float pred = red[0] + cb2[(size_t)ei * D_ + d];
        // sigmoid(pred)>0.5 <=> pred>0 ; cnt==0 diseases forced inactive
        flag[b * D_ + d] = (cnt[d] > 0.f && pred > 0.f) ? 1.f : 0.f;
    }
}

// routing weights wexp[b,s,e] = active/max(sum_active,1)
__global__ void routing_k(const float* __restrict__ flag, const float* __restrict__ maskf,
                          float* __restrict__ wexp)
{
    int idx = blockIdx.x * 256 + threadIdx.x;
    if (idx >= T_) return;
    int s = idx % S_, b = idx / S_;
    float a[E_];
    float sum = 0.f;
    for (int d = 0; d < D_; ++d) {
        float tm = (s == 0) ? 0.f : maskf[(s - 1) * D_ + d];
        float vv = flag[b * D_ + d] * tm;
        a[d] = vv; sum += vv;
    }
    a[D_] = 1.f; sum += 1.f;   // general expert always on
    float w = 1.0f / fmaxf(sum, 1.f);
    for (int e = 0; e < E_; ++e) wexp[(size_t)idx * E_ + e] = a[e] * w;
}

// out[e,t,r] = sum_k X[t,k] * W[e,r,k]    (rank-8 LoRA projection)
__global__ void small_proj_k(const float* __restrict__ X, const float* __restrict__ W,
                             float* __restrict__ out, int K)
{
    int idx = blockIdx.x * 256 + threadIdx.x;
    if (idx >= E_ * T_ * LR_) return;
    int r = idx % LR_;
    int et = idx / LR_;
    int t = et % T_, e = et / T_;
    const float* x = X + (size_t)t * K;
    const float* w = W + ((size_t)(e * LR_ + r)) * K;
    float acc = 0.f;
    for (int i = 0; i < K; ++i) acc += x[i] * w[i];
    out[idx] = acc;
}

// mdu[e,r,l] = sum_i ad[e,r,i] * bu[e,i,l]   (8x8 per expert)
__global__ void mdu_k(const float* __restrict__ ad, const float* __restrict__ bu,
                      float* __restrict__ mdu)
{
    int idx = blockIdx.x * 256 + threadIdx.x;
    if (idx >= E_ * LR_ * LR_) return;
    int l = idx % LR_;
    int er = idx / LR_;
    int r = er % LR_, e = er / LR_;
    const float* a = ad + ((size_t)(e * LR_ + r)) * I_;
    const float* b = bu + (size_t)e * I_ * LR_ + l;
    float acc = 0.f;
    for (int i = 0; i < I_; ++i) acc += a[i] * b[(size_t)i * LR_];
    mdu[idx] = acc;
}

// ============================================================
// Expert combine (one block per token):
// tmp2 = gA + SCAL * tmp1 @ Mdu^T ; o = base_out + SCAL*tmp2@Bd^T
// h = sum_e wexp_e * LN(o_e) + res1
// ============================================================
__global__ __launch_bounds__(256) void expert_combine_k(
    const float* __restrict__ base_out, const float* __restrict__ res1,
    const float* __restrict__ tmp1, const float* __restrict__ gA,
    const float* __restrict__ mdu, const float* __restrict__ bd,
    const float* __restrict__ wexp, float* __restrict__ hout)
{
    __shared__ float t2[LR_];
    __shared__ float red[256], red2[256];
    const int t = blockIdx.x;
    const int tid = threadIdx.x;
    float acc[3] = {0.f, 0.f, 0.f};
    float ov[3];
    const float* bo = base_out + (size_t)t * H_;

    for (int e = 0; e < E_; ++e) {
        const float we = wexp[(size_t)t * E_ + e];   // uniform across block
        if (we == 0.f) continue;
        if (tid < LR_) {
            const float* t1 = tmp1 + ((size_t)(e * T_ + t)) * LR_;
            const float* md = mdu + (size_t)(e * LR_ + tid) * LR_;
            float s = 0.f;
            #pragma unroll
            for (int l = 0; l < LR_; ++l) s += t1[l] * md[l];
            t2[tid] = gA[((size_t)(e * T_ + t)) * LR_ + tid] + SCAL_ * s;
        }
        __syncthreads();
        float s = 0.f, ss = 0.f;
        #pragma unroll
        for (int c = 0; c < 3; ++c) {
            int hh = tid + c * 256;
            const float* bdr = bd + ((size_t)e * H_ + hh) * LR_;
            float dsum = 0.f;
            #pragma unroll
            for (int r = 0; r < LR_; ++r) dsum += t2[r] * bdr[r];
            float o = bo[hh] + SCAL_ * dsum;
            ov[c] = o; s += o; ss += o * o;
        }
        red[tid] = s; red2[tid] = ss; __syncthreads();
        for (int st = 128; st > 0; st >>= 1) {
            if (tid < st) { red[tid] += red[tid + st]; red2[tid] += red2[tid + st]; }
            __syncthreads();
        }
        const float m   = red[0] / H_;
        const float var = red2[0] / H_ - m * m;
        const float inv = rsqrtf(var + 1e-5f);
        __syncthreads();
        #pragma unroll
        for (int c = 0; c < 3; ++c) acc[c] += we * (ov[c] - m) * inv;
    }
    #pragma unroll
    for (int c = 0; c < 3; ++c) {
        int hh = tid + c * 256;
        hout[(size_t)t * H_ + hh] = acc[c] + res1[(size_t)t * H_ + hh];
    }
}

// ============================================================
// Orchestration
// ============================================================
extern "C" void kernel_launch(void* const* d_in, const int* in_sizes, int n_in,
                              void* d_out, int out_size, void* d_ws, size_t ws_size,
                              hipStream_t stream)
{
    const float* region = (const float*)d_in[0];
    const float* cls    = (const float*)d_in[1];
    const float* ln1_g  = (const float*)d_in[2];
    const float* ln1_b  = (const float*)d_in[3];
    const float* ln2_g  = (const float*)d_in[4];
    const float* ln2_b  = (const float*)d_in[5];
    const float* wq = (const float*)d_in[6];   const float* bq = (const float*)d_in[7];
    const float* wk = (const float*)d_in[8];   const float* bk = (const float*)d_in[9];
    const float* wv = (const float*)d_in[10];  const float* bv = (const float*)d_in[11];
    const float* wo = (const float*)d_in[12];  const float* bo = (const float*)d_in[13];
    const float* wi = (const float*)d_in[14];  const float* bi = (const float*)d_in[15];
    const float* wof = (const float*)d_in[16]; const float* bof = (const float*)d_in[17];
    const float* lnf_g = (const float*)d_in[18]; const float* lnf_b = (const float*)d_in[19];
    const float* au = (const float*)d_in[20];  const float* bu = (const float*)d_in[21];
    const float* ad = (const float*)d_in[22];  const float* bd = (const float*)d_in[23];
    const float* cw1 = (const float*)d_in[24]; const float* cb1 = (const float*)d_in[25];
    const float* clg = (const float*)d_in[26]; const float* clb = (const float*)d_in[27];
    const float* cw2 = (const float*)d_in[28]; const float* cb2 = (const float*)d_in[29];
    const float* mask = (const float*)d_in[30];
    (void)in_sizes; (void)n_in; (void)out_size; (void)ws_size;

    float* ws = (float*)d_ws;
    size_t off = 0;
    auto alloc = [&](size_t n) { float* p = ws + off; off += n; return p; };
    const size_t TH = (size_t)T_ * H_;

    float* h    = alloc(TH);
    float* nx   = alloc(TH);               // also reused as attention ctx
    float* nres = alloc(TH);
    float* qb   = alloc(TH);               // reused as att (post-wo)
    float* kb   = alloc(TH);               // reused as res1
    float* vb   = alloc(TH);               // reused as base_out
    float* bhid = alloc((size_t)T_ * I_);
    float* pooled = alloc((size_t)B_ * D_ * H_);
    float* tmp1 = alloc((size_t)E_ * T_ * LR_);
    float* gAb  = alloc((size_t)E_ * T_ * LR_);
    float* mduB = alloc((size_t)E_ * LR_ * LR_);
    float* maskf = alloc(R_ * D_);
    float* cnt  = alloc(D_);
    float* flag = alloc(B_ * D_);
    float* wexp = alloc((size_t)T_ * E_);

    float* ctx  = nx;    // safe alias: attn reads q/k/v; ln1 output consumed by QKV gemms
    float* att  = qb;    // safe alias: wo-gemm reads ctx only
    float* res1 = kb;
    float* bout = vb;

    const int gTH = (int)((TH + 255) / 256);
    init_h_k<<<gTH, 256, 0, stream>>>(region, cls, h);
    mask_prep_k<<<1, 512, 0, stream>>>(mask, maskf, cnt);

    int ei = 0;
    for (int i = 0; i < L_; ++i) {
        layernorm_k<<<T_, 256, 0, stream>>>(h, ln1_g + i * H_, ln1_b + i * H_, nx, H_, 1e-12f);
        dim3 gq(H_ / 64, T_ / 128);
        gemm_wmma<0><<<gq, 256, 0, stream>>>(nx, wq + (size_t)i * H_ * H_, bq + i * H_, qb, T_, H_, H_);
        gemm_wmma<0><<<gq, 256, 0, stream>>>(nx, wk + (size_t)i * H_ * H_, bk + i * H_, kb, T_, H_, H_);
        gemm_wmma<0><<<gq, 256, 0, stream>>>(nx, wv + (size_t)i * H_ * H_, bv + i * H_, vb, T_, H_, H_);
        attn_k<<<B_ * NH_, 128, 0, stream>>>(qb, kb, vb, ctx);
        gemm_wmma<0><<<gq, 256, 0, stream>>>(ctx, wo + (size_t)i * H_ * H_, bo + i * H_, att, T_, H_, H_);
        add_k<<<gTH, 256, 0, stream>>>(att, h, res1, (int)TH);
        layernorm_k<<<T_, 256, 0, stream>>>(res1, ln2_g + i * H_, ln2_b + i * H_, nres, H_, 1e-12f);
        dim3 gi(I_ / 64, T_ / 128);
        gemm_wmma<1><<<gi, 256, 0, stream>>>(nres, wi + (size_t)i * I_ * H_, bi + i * I_, bhid, T_, I_, H_);
        gemm_wmma<0><<<gq, 256, 0, stream>>>(bhid, wof + (size_t)i * H_ * I_, bof + i * H_, bout, T_, H_, I_);

        if ((i & 1) == 0) {
            pool_k<<<(B_ * D_ * H_ + 255) / 256, 256, 0, stream>>>(att, maskf, cnt, pooled);
            classify_k<<<B_ * D_, 128, 0, stream>>>(pooled, cw1, cb1, clg, clb, cw2, cb2, cnt, flag, ei);
            routing_k<<<(T_ + 255) / 256, 256, 0, stream>>>(flag, maskf, wexp);
            small_proj_k<<<(E_ * T_ * LR_ + 255) / 256, 256, 0, stream>>>(
                nres, au + (size_t)ei * E_ * LR_ * H_, tmp1, H_);
            small_proj_k<<<(E_ * T_ * LR_ + 255) / 256, 256, 0, stream>>>(
                bhid, ad + (size_t)ei * E_ * LR_ * I_, gAb, I_);
            mdu_k<<<(E_ * LR_ * LR_ + 255) / 256, 256, 0, stream>>>(
                ad + (size_t)ei * E_ * LR_ * I_, bu + (size_t)ei * E_ * I_ * LR_, mduB);
            expert_combine_k<<<T_, 256, 0, stream>>>(
                bout, res1, tmp1, gAb, mduB, bd + (size_t)ei * E_ * H_ * LR_, wexp, h);
            ++ei;
        } else {
            add_k<<<gTH, 256, 0, stream>>>(res1, bout, h, (int)TH);
        }
    }
    layernorm_k<<<T_, 256, 0, stream>>>(h, lnf_g, lnf_b, (float*)d_out, H_, 1e-12f);
}
